// Net_19748259627141
// MI455X (gfx1250) — matrix operand order repaired and checked by reading, not compiled, not run on hardware
//
#include <hip/hip_runtime.h>

typedef __attribute__((ext_vector_type(16))) _Float16 v16h;
typedef __attribute__((ext_vector_type(8)))  _Float16 h8;
typedef __attribute__((ext_vector_type(8)))  float    v8f;
typedef __attribute__((ext_vector_type(4)))  int      v4i;

#define T_STEPS 250
#define BATCH   256
#define NI      700
#define NH      1024
#define NO      20
#define MROWS   (T_STEPS * BATCH)   /* 64000 */
#define KP      704                 /* K padded to 22*32 for GEMM1 */
#define LDT     40                  /* LDS K-stride in halves: 80B, 16B-aligned, bank-friendly */

#define GLOBAL_AS __attribute__((address_space(1)))
#define LDS_AS    __attribute__((address_space(3)))

#if __has_builtin(__builtin_amdgcn_global_load_async_to_lds_b128)
#define HAS_ASYNC_LDS 1
static __device__ __forceinline__ void async_b128(const _Float16* g, _Float16* l) {
  GLOBAL_AS void* gv = (GLOBAL_AS void*)g;
  LDS_AS   void* lv = (LDS_AS void*)l;
  __builtin_amdgcn_global_load_async_to_lds_b128((GLOBAL_AS v4i*)gv, (LDS_AS v4i*)lv, 0, 0);
}
#else
#define HAS_ASYNC_LDS 0
static __device__ __forceinline__ void async_b128(const _Float16* g, _Float16* l) {
  *(uint4*)l = *(const uint4*)g;   // sync fallback: b128 load + b128 ds store
}
#endif

#if __has_builtin(__builtin_amdgcn_s_wait_asynccnt)
#define WAIT_ASYNC(n) __builtin_amdgcn_s_wait_asynccnt(n)
#else
#define WAIT_ASYNC(n) asm volatile("s_wait_asynccnt %0" :: "i"(n))
#endif

static __device__ __forceinline__ v16h load_frag16(const _Float16* p_lo, const _Float16* p_hi) {
  h8 lo = *(const h8*)p_lo;
  h8 hi = *(const h8*)p_hi;
  return __builtin_shufflevector(lo, hi, 0,1,2,3,4,5,6,7,8,9,10,11,12,13,14,15);
}

// ---------------------------------------------------------------------------
// One-time conversions (hoists all f32->f16 VALU work out of the GEMM loop).
// spikes: f32 [M][700] -> f16 [M][704] zero-padded (exact: values are {0,1})
// ---------------------------------------------------------------------------
__global__ __launch_bounds__(256) void snn_cvt_spikes(const float* __restrict__ X,
                                                      _Float16* __restrict__ Y)
{
  int i = blockIdx.x * 256 + threadIdx.x;        // MROWS*KP/8 = 5,632,000 chunks
  int row = i / (KP / 8);
  int k0  = (i % (KP / 8)) * 8;
  h8 o;
#pragma unroll
  for (int j = 0; j < 8; ++j) {
    int k = k0 + j;
    float v = 0.f;
    if (k < NI) v = X[(size_t)row * NI + k];
    o[j] = (_Float16)v;
  }
  *(h8*)(Y + (size_t)row * KP + k0) = o;
}

// W1: f32 [NH][700] -> hi/lo f16 planes [NH][704] (split => fp32-level accuracy)
__global__ __launch_bounds__(256) void snn_cvt_w1(const float* __restrict__ W,
                                                  _Float16* __restrict__ Wh,
                                                  _Float16* __restrict__ Wl)
{
  int i = blockIdx.x * 256 + threadIdx.x;        // NH*KP/8 = 90,112 chunks
  int row = i / (KP / 8);
  int k0  = (i % (KP / 8)) * 8;
  h8 oh, ol;
#pragma unroll
  for (int j = 0; j < 8; ++j) {
    int k = k0 + j;
    float v = 0.f;
    if (k < NI) v = W[(size_t)row * NI + k];
    _Float16 h = (_Float16)v;
    oh[j] = h;
    ol[j] = (_Float16)(v - (float)h);
  }
  *(h8*)(Wh + (size_t)row * KP + k0) = oh;
  *(h8*)(Wl + (size_t)row * KP + k0) = ol;
}

// ---------------------------------------------------------------------------
// GEMM1: cur1[M][NH] = spikes_h[M][704] * (W1h+W1l)[NH][704]^T
// Block 128x128, K-step 32, 8 waves = 4(M)x2(N), 2x4 accums/wave, two WMMAs
// (hi+lo W) per tile. Double-buffered LDS fed by async global->LDS b128 DMA
// (ASYNCcnt-tracked), so the WMMA stream overlaps the next tile's fill.
// ---------------------------------------------------------------------------
__global__ __launch_bounds__(256) void snn_gemm1(const _Float16* __restrict__ Sh,
                                                 const _Float16* __restrict__ W1h,
                                                 const _Float16* __restrict__ W1l,
                                                 float* __restrict__ C)
{
  __shared__ _Float16 As[2][128 * LDT];
  __shared__ _Float16 Bh[2][128 * LDT];
  __shared__ _Float16 Bl[2][128 * LDT];

  const int tid  = threadIdx.x;
  const int lane = tid & 31;
  const int wave = tid >> 5;
  const int wm   = wave & 3;
  const int wn   = wave >> 2;
  const int m0   = blockIdx.y * 128;     // 500 M panels
  const int n0   = blockIdx.x * 128;     // 8 N panels (adjacent blocks share A panel in L2)

  v8f acc[2][4];
#pragma unroll
  for (int i = 0; i < 2; ++i)
#pragma unroll
    for (int j = 0; j < 4; ++j)
      acc[i][j] = (v8f){0.f,0.f,0.f,0.f,0.f,0.f,0.f,0.f};

  const int arow = wm * 32 + (lane & 15);
  const int akb  = (lane < 16) ? 0 : 16;
  const int bcol = wn * 64 + (lane & 15);
  const int bkb  = (lane < 16) ? 0 : 16;

  // issue one 128x32-half tile triple: 6 x b128 per thread (2 A + 2 Bh + 2 Bl)
  auto issue = [&](int buf, int k0) {
#pragma unroll
    for (int r = 0; r < 2; ++r) {
      int idx = r * 256 + tid;           // 512 chunks of 8 halves per tile
      int row = idx >> 2;
      int seg = (idx & 3) * 8;
      async_b128(Sh  + (size_t)(m0 + row) * KP + k0 + seg, &As[buf][row * LDT + seg]);
      async_b128(W1h + (size_t)(n0 + row) * KP + k0 + seg, &Bh[buf][row * LDT + seg]);
      async_b128(W1l + (size_t)(n0 + row) * KP + k0 + seg, &Bl[buf][row * LDT + seg]);
    }
  };

  const int NK = KP / 32;                // 22
  issue(0, 0);
  for (int it = 0; it < NK; ++it) {
    const int buf = it & 1;
    if (it + 1 < NK) {
      issue(buf ^ 1, (it + 1) * 32);
#if HAS_ASYNC_LDS
      WAIT_ASYNC(6);                     // tile `it` landed; only next tile outstanding
#endif
    } else {
#if HAS_ASYNC_LDS
      WAIT_ASYNC(0);
#endif
    }
    __syncthreads();                     // all waves' tile-`it` data visible

    v16h af[2];
#pragma unroll
    for (int i = 0; i < 2; ++i) {
      const _Float16* p = &As[buf][(arow + i * 16) * LDT + akb];
      af[i] = load_frag16(p, p + 8);
    }
#pragma unroll
    for (int j = 0; j < 4; ++j) {
      const _Float16* ph = &Bh[buf][(bcol + j * 16) * LDT + bkb];
      const _Float16* pl = &Bl[buf][(bcol + j * 16) * LDT + bkb];
      v16h bhf = load_frag16(ph, ph + 8);
      v16h blf = load_frag16(pl, pl + 8);
#pragma unroll
      for (int i = 0; i < 2; ++i) {
        acc[i][j] = __builtin_amdgcn_wmma_f32_16x16x32_f16(false, af[i], false, bhf,
                                                           (short)0, acc[i][j], false, false);
        acc[i][j] = __builtin_amdgcn_wmma_f32_16x16x32_f16(false, af[i], false, blf,
                                                           (short)0, acc[i][j], false, false);
      }
    }
    __syncthreads();                     // buf may be overwritten by iter it+2's issue
  }

  const int rbase = m0 + wm * 32 + ((lane >= 16) ? 8 : 0);
  const int cbase = n0 + wn * 64 + (lane & 15);
#pragma unroll
  for (int i = 0; i < 2; ++i)
#pragma unroll
    for (int j = 0; j < 4; ++j)
#pragma unroll
      for (int v = 0; v < 8; ++v)
        C[(size_t)(rbase + i * 16 + v) * NH + cbase + j * 16] = acc[i][j][v];
}

// ---------------------------------------------------------------------------
// LIF scan: one thread per (b,h) neuron, coalesced over B*NH, sequential in T.
// ---------------------------------------------------------------------------
__global__ __launch_bounds__(256) void snn_lif(const float* __restrict__ cur1,
                                               _Float16* __restrict__ spk1)
{
  const int j = blockIdx.x * blockDim.x + threadIdx.x;
  const size_t stride = (size_t)BATCH * NH;
  float mem = 0.f;
#pragma unroll 1
  for (int t = 0; t < T_STEPS; ++t) {
    float c = cur1[(size_t)t * stride + j];
    float reset = (mem > 1.0f) ? 1.0f : 0.0f;            // heaviside(mem_prev - thr)
    mem = 0.9f * mem + c - reset;                        // reset_mechanism='subtract'
    spk1[(size_t)t * stride + j] = (mem > 1.0f) ? (_Float16)1.0f : (_Float16)0.0f;
  }
}

// ---------------------------------------------------------------------------
// GEMM2: cur2[M][NO] = spk1[M][1024] * W2[20][1024]^T, N padded 20->32.
// Block 256x32, 8 waves along M, 2x2 accums/wave. Async-DMA'd, double-
// buffered A tiles (spk1 already f16); small W2 tile converted in-flight.
// ---------------------------------------------------------------------------
__global__ __launch_bounds__(256) void snn_gemm2(const _Float16* __restrict__ S,
                                                 const float* __restrict__ W2,
                                                 float* __restrict__ C2)
{
  __shared__ _Float16 As[2][256 * LDT];
  __shared__ _Float16 Bs[2][32 * LDT];

  const int tid  = threadIdx.x;
  const int lane = tid & 31;
  const int wave = tid >> 5;
  const int m0   = blockIdx.x * 256;

  v8f acc[2][2];
#pragma unroll
  for (int i = 0; i < 2; ++i)
#pragma unroll
    for (int j = 0; j < 2; ++j)
      acc[i][j] = (v8f){0.f,0.f,0.f,0.f,0.f,0.f,0.f,0.f};

  const int arow = wave * 32 + (lane & 15);
  const int akb  = (lane < 16) ? 0 : 16;
  const int bcol = lane & 15;
  const int bkb  = (lane < 16) ? 0 : 16;
  const int bn   = tid >> 3;             // 0..31
  const int bko  = (tid & 7) * 4;        // 0..28

  auto issue = [&](int buf, int k0) {
#pragma unroll
    for (int r = 0; r < 4; ++r) {        // A: 1024 chunks of 8 halves, 4/thread
      int idx = r * 256 + tid;
      int row = idx >> 2;
      int seg = (idx & 3) * 8;
      async_b128(S + (size_t)(m0 + row) * NH + k0 + seg, &As[buf][row * LDT + seg]);
    }
    float4 v = {0.f, 0.f, 0.f, 0.f};     // B: rows 20..31 stay zero
    if (bn < NO) v = *(const float4*)(W2 + (size_t)bn * NH + k0 + bko);
    _Float16* d = &Bs[buf][bn * LDT + bko];
    d[0] = (_Float16)v.x; d[1] = (_Float16)v.y; d[2] = (_Float16)v.z; d[3] = (_Float16)v.w;
  };

  const int NK = NH / 32;                // 32
  issue(0, 0);
  for (int it = 0; it < NK; ++it) {
    const int buf = it & 1;
    if (it + 1 < NK) {
      issue(buf ^ 1, (it + 1) * 32);
#if HAS_ASYNC_LDS
      WAIT_ASYNC(4);
#endif
    } else {
#if HAS_ASYNC_LDS
      WAIT_ASYNC(0);
#endif
    }
    __syncthreads();

    v16h af[2], bf[2];
#pragma unroll
    for (int i = 0; i < 2; ++i) {
      const _Float16* p = &As[buf][(arow + i * 16) * LDT + akb];
      af[i] = load_frag16(p, p + 8);
    }
#pragma unroll
    for (int j = 0; j < 2; ++j) {
      const _Float16* p = &Bs[buf][(bcol + j * 16) * LDT + bkb];
      bf[j] = load_frag16(p, p + 8);
    }
#pragma unroll
    for (int i = 0; i < 2; ++i)
#pragma unroll
      for (int j = 0; j < 2; ++j)
        acc[i][j] = __builtin_amdgcn_wmma_f32_16x16x32_f16(false, af[i], false, bf[j],
                                                           (short)0, acc[i][j], false, false);
    __syncthreads();
  }

  const int rbase = m0 + wave * 32 + ((lane >= 16) ? 8 : 0);
#pragma unroll
  for (int i = 0; i < 2; ++i)
#pragma unroll
    for (int j = 0; j < 2; ++j) {
      const int col = j * 16 + (lane & 15);
      if (col < NO) {
#pragma unroll
        for (int v = 0; v < 8; ++v)
          C2[(size_t)(rbase + i * 16 + v) * NO + col] = acc[i][j][v];
      }
    }
}

// ---------------------------------------------------------------------------
// Readout leak + softmax: wave per batch element, lane = class, wave32
// shuffle reductions over the 20 active lanes.
// ---------------------------------------------------------------------------
__global__ __launch_bounds__(256) void snn_readout(const float* __restrict__ cur2,
                                                   float* __restrict__ out_mem,
                                                   float* __restrict__ out_sm)
{
  const int lane = threadIdx.x & 31;
  const int wave = threadIdx.x >> 5;
  const int b    = blockIdx.x * 8 + wave;
  const bool act = (lane < NO);
  float mem = 0.f;
#pragma unroll 1
  for (int t = 0; t < T_STEPS; ++t) {
    const size_t base = ((size_t)t * BATCH + b) * NO;
    float c = act ? cur2[base + lane] : 0.f;
    mem = 0.9f * mem + c;                    // reset_mechanism='none'
    float mx = act ? mem : -3.0e38f;
#pragma unroll
    for (int off = 16; off >= 1; off >>= 1) mx = fmaxf(mx, __shfl_xor(mx, off, 32));
    float e = act ? __expf(mem - mx) : 0.f;
    float s = e;
#pragma unroll
    for (int off = 16; off >= 1; off >>= 1) s += __shfl_xor(s, off, 32);
    if (act) {
      out_mem[base + lane] = mem;
      out_sm[base + lane]  = e / s;
    }
  }
}

extern "C" void kernel_launch(void* const* d_in, const int* in_sizes, int n_in,
                              void* d_out, int out_size, void* d_ws, size_t ws_size,
                              hipStream_t stream)
{
  const float* spikes = (const float*)d_in[0];  // [250][256][700]
  const float* W1     = (const float*)d_in[1];  // [1024][700]
  const float* W2     = (const float*)d_in[2];  // [20][1024]
  float* out = (float*)d_out;

  // workspace layout (bytes):
  //   cur1    f32 [M][NH]   262,144,000
  //   spk1    f16 [M][NH]   131,072,000
  //   cur2    f32 [M][NO]     5,120,000
  //   spikesh f16 [M][704]   90,112,000
  //   W1h/W1l f16 [NH][704]   1,441,792 each        total ~491.3 MB
  char* ws = (char*)d_ws;
  float*    cur1    = (float*)ws;
  _Float16* spk1    = (_Float16*)(ws + 262144000ull);
  float*    cur2    = (float*)   (ws + 393216000ull);
  _Float16* spikesh = (_Float16*)(ws + 398336000ull);
  _Float16* W1h     = (_Float16*)(ws + 488448000ull);
  _Float16* W1l     = (_Float16*)(ws + 489889792ull);

  snn_cvt_spikes<<<(MROWS * (KP / 8)) / 256, 256, 0, stream>>>(spikes, spikesh);
  snn_cvt_w1<<<(NH * (KP / 8)) / 256, 256, 0, stream>>>(W1, W1h, W1l);

  dim3 g1(NH / 128, MROWS / 128);   // x = N panels (share A panel), y = M panels
  snn_gemm1<<<g1, 256, 0, stream>>>(spikesh, W1h, W1l, cur1);
  snn_lif<<<(BATCH * NH) / 256, 256, 0, stream>>>(cur1, spk1);
  snn_gemm2<<<MROWS / 256, 256, 0, stream>>>(spk1, W2, cur2);

  float* out_mem = out;
  float* out_sm  = out + (size_t)T_STEPS * BATCH * NO;
  snn_readout<<<BATCH / 8, 256, 0, stream>>>(cur2, out_mem, out_sm);

  (void)in_sizes; (void)n_in; (void)out_size; (void)ws_size;
}
